// RotationEstimateNN_21191368639249
// MI455X (gfx1250) — matrix operand order repaired and checked by reading, not compiled, not run on hardware
//
#include <hip/hip_runtime.h>
#include <math.h>

typedef __attribute__((ext_vector_type(16))) _Float16 v16h;
typedef __attribute__((ext_vector_type(8)))  _Float16 v8h;
typedef __attribute__((ext_vector_type(8)))  float    v8f;

#define CDIV(a,b) (((a)+(b)-1)/(b))
#define USE_ASYNC_LDS 1

constexpr int Bc  = 32;
constexpr int P0  = 1024;
constexpr int P1c = 384;
constexpr int P2c = 128;

// ----------------------------------------------------------------------------
// CDNA5 async global->LDS copy (ASYNCcnt-tracked, no VGPR data movement)
// ----------------------------------------------------------------------------
__device__ __forceinline__ void async_copy_f32(float* lds_dst, const float* gsrc)
{
#if USE_ASYNC_LDS
  // low 32 bits of the flat LDS-aperture address select the LDS offset
  unsigned l = (unsigned)(size_t)lds_dst;
  asm volatile("global_load_async_to_lds_b32 %0, %1, off"
               :: "v"(l), "v"(gsrc) : "memory");
#else
  *lds_dst = *gsrc;
#endif
}
__device__ __forceinline__ void async_join()
{
#if USE_ASYNC_LDS
  asm volatile("s_wait_asynccnt 0x0" ::: "memory");
#endif
}

// ============================================================================
// WMMA GEMM:  C[M,N] = act(A[M,K] @ W[K,N] + bias[N])
// f32 in/out, f16 multiply via v_wmma_f32_16x16x32_f16, f32 accumulate.
// 128 threads = 4 waves per block; block computes a 32x32 C tile, each wave
// one 16x16 WMMA tile. B tile stored TRANSPOSED in LDS so both fragments are
// loaded as contiguous 16B vectors (ds_load_b128), no scalar packing.
// ACT: 0 = none, 1 = relu, 2 = elu
// ============================================================================
template<int ACT>
__global__ __launch_bounds__(128)
void wmma_gemm_kernel(const float* __restrict__ A, const float* __restrict__ W,
                      const float* __restrict__ bias, float* __restrict__ C,
                      int M, int N, int K)
{
  __shared__ _Float16 sA [32 * 32];   // [m][k]
  __shared__ _Float16 sBt[32 * 32];   // [n][k]   (transposed B tile)
  const int tid  = threadIdx.x;
  const int lane = tid & 31;
  const int wv   = tid >> 5;               // wave id 0..3
  const int tm = blockIdx.y * 32;
  const int tn = blockIdx.x * 32;
  const int wm = (wv & 1) * 16;            // wave's row offset inside tile
  const int wn = (wv >> 1) * 16;           // wave's col offset inside tile
  const bool mn_full = (tm + 32 <= M) && (tn + 32 <= N);
  v8f acc = {};
  for (int k0 = 0; k0 < K; k0 += 32) {
    if (mn_full && (k0 + 32 <= K)) {
      // branch-free staging: thread -> row r = tid/4, cols c0..c0+7
      const int r  = tid >> 2;
      const int c0 = (tid & 3) << 3;
      const float* ap = A + (size_t)(tm + r) * K + k0 + c0;
      const float* wp = W + (size_t)(k0 + r) * N + tn + c0;
      #pragma unroll
      for (int j = 0; j < 8; ++j) sA[r * 32 + c0 + j] = (_Float16)ap[j];
      #pragma unroll
      for (int j = 0; j < 8; ++j) sBt[(c0 + j) * 32 + r] = (_Float16)wp[j];
      if (k0 + 32 < K) {   // global_prefetch_b8 of next tiles
        __builtin_prefetch(ap + 32, 0, 0);
        __builtin_prefetch(W + (size_t)(k0 + 32 + r) * N + tn + c0, 0, 0);
      }
    } else {
      for (int i = tid; i < 1024; i += 128) {
        int r = i >> 5, c = i & 31;
        int gm = tm + r, gk = k0 + c;
        sA[i] = (_Float16)((gm < M && gk < K) ? A[(size_t)gm * K + gk] : 0.f);
      }
      for (int i = tid; i < 1024; i += 128) {
        int r = i >> 5, c = i & 31;            // r = k row, c = n col
        int gk = k0 + r, gn = tn + c;
        sBt[c * 32 + r] = (_Float16)((gk < K && gn < N) ? W[(size_t)gk * N + gn] : 0.f);
      }
    }
    __syncthreads();
    // A fragment (16-bit A 16x32, ISA 7.12.2): lane half-group selects k-base;
    // each lane reads two contiguous 8-half (16B) runs.
    const int arow  = wm + (lane & 15);
    const int abase = (lane & 16) ? 8 : 0;
    v8h a_lo = *(const v8h*)&sA[arow * 32 + abase];
    v8h a_hi = *(const v8h*)&sA[arow * 32 + abase + 16];
    v16h af = __builtin_shufflevector(a_lo, a_hi,
                0,1,2,3,4,5,6,7,8,9,10,11,12,13,14,15);
    // B fragment: same k order as A; with sBt[n][k] this is two 8-half
    // runs at k = bko and bko + 16 -> two 16B vector loads.
    const int bcol = wn + (lane & 15);
    const int bko  = (lane & 16) ? 8 : 0;
    v8h b_lo = *(const v8h*)&sBt[bcol * 32 + bko];
    v8h b_hi = *(const v8h*)&sBt[bcol * 32 + bko + 16];
    v16h bf = __builtin_shufflevector(b_lo, b_hi,
                0,1,2,3,4,5,6,7,8,9,10,11,12,13,14,15);
    acc = __builtin_amdgcn_wmma_f32_16x16x32_f16(false, af, false, bf,
                                                 (short)0, acc, false, false);
    __syncthreads();
  }
  // C layout: lane l, VGPR r -> C[m = r + 8*(l/16)][n = l%16]
  const int n  = tn + wn + (lane & 15);
  const int m0 = tm + wm + ((lane >> 4) << 3);
  if (n < N) {
    float bv = bias ? bias[n] : 0.f;
    #pragma unroll
    for (int r = 0; r < 8; ++r) {
      int m = m0 + r;
      if (m < M) {
        float v = acc[r] + bv;
        if (ACT == 1) v = fmaxf(v, 0.f);
        if (ACT == 2) v = (v > 0.f) ? v : expm1f(v);
        C[(size_t)m * N + n] = v;
      }
    }
  }
}

// ============================================================================
// BatchNorm (training-mode batch statistics): stats + apply
// ============================================================================
__global__ __launch_bounds__(256)
void bn_stats_kernel(const float* __restrict__ X, float* __restrict__ mean,
                     float* __restrict__ var, long Nrows, int C)
{
  const int c = blockIdx.x;
  float s = 0.f, s2 = 0.f;
  for (long i = threadIdx.x; i < Nrows; i += 256) {
    float v = X[i * C + c];
    s += v; s2 += v * v;
  }
  __shared__ float rs[256], rs2[256];
  rs[threadIdx.x] = s; rs2[threadIdx.x] = s2;
  __syncthreads();
  for (int o = 128; o > 0; o >>= 1) {
    if (threadIdx.x < o) {
      rs[threadIdx.x]  += rs[threadIdx.x + o];
      rs2[threadIdx.x] += rs2[threadIdx.x + o];
    }
    __syncthreads();
  }
  if (threadIdx.x == 0) {
    float m = rs[0] / (float)Nrows;
    mean[c] = m;
    var[c]  = rs2[0] / (float)Nrows - m * m;
  }
}

__global__ void bn_apply_kernel(float* __restrict__ X,
                                const float* __restrict__ mean,
                                const float* __restrict__ var,
                                const float* __restrict__ g,
                                const float* __restrict__ be,
                                long Nrows, int C)
{
  long i = (long)blockIdx.x * blockDim.x + threadIdx.x;
  if (i >= Nrows * C) return;
  int c = (int)(i % C);
  X[i] = (X[i] - mean[c]) * rsqrtf(var[c] + 1e-5f) * g[c] + be[c];
}

// ============================================================================
// Brute-force kNN (self included, ties -> lowest index), cloud cached in LDS
// via CDNA5 async global->LDS loads
// ============================================================================
template<int KN>
__global__ __launch_bounds__(256)
void knn_kernel(const float* __restrict__ pos, int* __restrict__ nbr, int Pn)
{
  const int b = blockIdx.x;
  __shared__ float sp[P0 * 3];
  {
    const float* src = pos + (size_t)b * Pn * 3;
    for (int i = threadIdx.x; i < Pn * 3; i += 256)
      async_copy_f32(&sp[i], src + i);
    async_join();
  }
  __syncthreads();
  for (int q = threadIdx.x; q < Pn; q += 256) {
    const float qx = sp[q * 3], qy = sp[q * 3 + 1], qz = sp[q * 3 + 2];
    float bd[KN]; int bi[KN];
    #pragma unroll
    for (int k = 0; k < KN; ++k) { bd[k] = 3.0e38f; bi[k] = 0; }
    for (int j = 0; j < Pn; ++j) {
      float dx = sp[j * 3] - qx, dy = sp[j * 3 + 1] - qy, dz = sp[j * 3 + 2] - qz;
      float d = dx * dx + dy * dy + dz * dz;
      if (d < bd[KN - 1]) {
        int k = KN - 1;
        while (k > 0 && bd[k - 1] > d) { bd[k] = bd[k - 1]; bi[k] = bi[k - 1]; --k; }
        bd[k] = d; bi[k] = j;
      }
    }
    for (int k = 0; k < KN; ++k)
      nbr[((size_t)b * Pn + q) * KN + k] = bi[k];
  }
}

// random-dilation subsample: pick K of K*dil neighbors (hash-based PRNG)
__global__ void dilate_kernel(const int* __restrict__ nbrF, int* __restrict__ nbr,
                              long BPK, int K, int Kd, unsigned seed)
{
  long i = (long)blockIdx.x * blockDim.x + threadIdx.x;
  if (i >= BPK) return;
  long bp = i / K; int k = (int)(i % K);
  unsigned h = (unsigned)bp * 2654435761u ^ ((unsigned)k * 40503u) ^ seed;
  h ^= h >> 13; h *= 0x5bd1e995u; h ^= h >> 15;
  nbr[i] = nbrF[bp * Kd + (int)(h % (unsigned)Kd)];
}

// rel[b,p,k,:] = pos[b,nbr] - pos[b,p]
__global__ void gather_rel_kernel(const float* __restrict__ pos,
                                  const int* __restrict__ nbr,
                                  float* __restrict__ rel, int Pn, int K)
{
  long i = (long)blockIdx.x * blockDim.x + threadIdx.x;
  long BPK = (long)Bc * Pn * K;
  if (i >= BPK) return;
  long bp = i / K;
  int  b  = (int)(bp / Pn);
  int  j  = nbr[i];
  const float* pj = pos + ((size_t)b * Pn + j) * 3;
  const float* pp = pos + (size_t)bp * 3;
  rel[i * 3 + 0] = pj[0] - pp[0];
  rel[i * 3 + 1] = pj[1] - pp[1];
  rel[i * 3 + 2] = pj[2] - pp[2];
}

// xcat[b,p,k,:Cd]=xs ; xcat[b,p,k,Cd:]=x[b,nbr[b,p,k],:]
__global__ void concat_kernel(const float* __restrict__ xs,
                              const float* __restrict__ x,
                              const int* __restrict__ nbr,
                              float* __restrict__ xcat,
                              int Pn, int K, int Cd, int Cin)
{
  const int Cm = Cd + Cin;
  long i = (long)blockIdx.x * blockDim.x + threadIdx.x;
  long tot = (long)Bc * Pn * K * Cm;
  if (i >= tot) return;
  int  c   = (int)(i % Cm);
  long bpk = i / Cm;
  if (c < Cd) {
    xcat[i] = xs[bpk * Cd + c];
  } else {
    long bp = bpk / K;
    int  b  = (int)(bp / Pn);
    int  j  = nbr[bpk];
    xcat[i] = x[((size_t)b * Pn + j) * Cin + (c - Cd)];
  }
}

// X-transform conv: out[bp,g,j] = act( sum_t in[bp,g,t] * w[g,j,t] + b[g*K+j] )
__global__ void tconv_kernel(const float* __restrict__ tin,
                             const float* __restrict__ w,
                             const float* __restrict__ bias,
                             float* __restrict__ tout, long BP, int K, int elu)
{
  long i = (long)blockIdx.x * blockDim.x + threadIdx.x;
  long tot = BP * K * K;
  if (i >= tot) return;
  int  j  = (int)(i % K);
  long t1 = i / K;
  int  g  = (int)(t1 % K);
  long bp = t1 / K;
  const float* tr = tin + bp * K * K + (long)g * K;
  const float* wr = w + ((long)g * K + j) * K;
  float s = bias[g * K + j];
  for (int t = 0; t < K; ++t) s += tr[t] * wr[t];
  if (elu) s = (s > 0.f) ? s : expm1f(s);
  tout[i] = s;
}

// xt[c,j] = sum_k xs[k,c]*T[k,j]; y[c*dm+m] = sum_j xt[c,j]*dww[c,m,j] + dwb
__global__ void xt_dw_kernel(const float* __restrict__ xs,  // [BP,K,Cm]
                             const float* __restrict__ T,   // [BP,K,K]
                             const float* __restrict__ dww, // [Cm,dm,K]
                             const float* __restrict__ dwb, // [Cm*dm]
                             float* __restrict__ y,         // [BP,Cm*dm]
                             long BP, int K, int Cm, int dm)
{
  long i = (long)blockIdx.x * blockDim.x + threadIdx.x;
  if (i >= BP * Cm) return;
  int  c  = (int)(i % Cm);
  long bp = i / Cm;
  const float* xsp = xs + bp * (long)K * Cm;
  const float* Tp  = T + bp * (long)K * K;
  float xt[16];
  for (int j = 0; j < K; ++j) {
    float s = 0.f;
    for (int k = 0; k < K; ++k) s += xsp[(long)k * Cm + c] * Tp[k * K + j];
    xt[j] = s;
  }
  for (int m = 0; m < dm; ++m) {
    float s = dwb[c * dm + m];
    const float* wr = dww + ((long)c * dm + m) * K;
    for (int j = 0; j < K; ++j) s += xt[j] * wr[j];
    y[bp * (long)(Cm * dm) + c * dm + m] = s;
  }
}

// ============================================================================
// Farthest point sampling: one workgroup per cloud, everything in LDS
// ============================================================================
__global__ __launch_bounds__(256)
void fps_kernel(const float* __restrict__ pos, int* __restrict__ idx, int Pn, int M)
{
  const int b = blockIdx.x;
  __shared__ float sp[P0 * 3];
  __shared__ float dmin[P0];
  __shared__ float rmax[256];
  __shared__ int   rarg[256];
  __shared__ int   scur;
  {
    const float* src = pos + (size_t)b * Pn * 3;
    for (int i = threadIdx.x; i < Pn * 3; i += 256)
      async_copy_f32(&sp[i], src + i);
    async_join();
  }
  for (int i = threadIdx.x; i < Pn; i += 256) dmin[i] = 1e10f;
  if (threadIdx.x == 0) scur = 0;
  __syncthreads();
  for (int m = 0; m < M; ++m) {
    const int cur = scur;
    const float cx = sp[cur * 3], cy = sp[cur * 3 + 1], cz = sp[cur * 3 + 2];
    float best = -1.f; int barg = 0;
    for (int i = threadIdx.x; i < Pn; i += 256) {
      float dx = sp[i * 3] - cx, dy = sp[i * 3 + 1] - cy, dz = sp[i * 3 + 2] - cz;
      float nd = fminf(dmin[i], dx * dx + dy * dy + dz * dz);
      dmin[i] = nd;
      if (nd > best) { best = nd; barg = i; }
    }
    rmax[threadIdx.x] = best; rarg[threadIdx.x] = barg;
    __syncthreads();
    for (int o = 128; o > 0; o >>= 1) {
      if (threadIdx.x < o) {
        bool take = (rmax[threadIdx.x + o] > rmax[threadIdx.x]) ||
                    (rmax[threadIdx.x + o] == rmax[threadIdx.x] &&
                     rarg[threadIdx.x + o] < rarg[threadIdx.x]);
        if (take) { rmax[threadIdx.x] = rmax[threadIdx.x + o];
                    rarg[threadIdx.x] = rarg[threadIdx.x + o]; }
      }
      __syncthreads();
    }
    if (threadIdx.x == 0) { idx[(size_t)b * M + m] = cur; scur = rarg[0]; }
    __syncthreads();
  }
}

// out[b,m,c] = x[b, idx[b,m], c]
__global__ void take_kernel(const float* __restrict__ x, const int* __restrict__ idx,
                            float* __restrict__ out, int Pn, int M, int C)
{
  long i = (long)blockIdx.x * blockDim.x + threadIdx.x;
  long tot = (long)Bc * M * C;
  if (i >= tot) return;
  int  c = (int)(i % C);
  long t = i / C;
  int  m = (int)(t % M);
  int  b = (int)(t / M);
  out[i] = x[((size_t)b * Pn + idx[b * M + m]) * C + c];
}

// ============================================================================
// Small head kernels
// ============================================================================
__global__ void meanpos_kernel(const float* __restrict__ pos, float* __restrict__ mp)
{
  int i = blockIdx.x * blockDim.x + threadIdx.x;
  if (i >= Bc * 3) return;
  int b = i / 3, d = i % 3;
  float s = 0.f;
  for (int p = 0; p < P0; ++p) s += pos[((size_t)b * P0 + p) * 3 + d];
  mp[i] = s / (float)P0;
}

__global__ void cls_kernel(const float* __restrict__ mp, const float* __restrict__ clsw,
                           float* __restrict__ clsb)
{
  int b = blockIdx.x * blockDim.x + threadIdx.x;
  if (b >= Bc) return;
  float best = -3.0e38f; int arg = 0;
  for (int k = 0; k < 8; ++k) {
    float s = 0.f;
    for (int d = 0; d < 3; ++d) s += mp[b * 3 + d] * clsw[d * 8 + k];
    if (s > best) { best = s; arg = k; }
  }
  for (int k = 0; k < 8; ++k) clsb[b * 8 + k] = (k == arg) ? 1.f : 0.f;
}

__global__ void meanpool_kernel(const float* __restrict__ x, float* __restrict__ feat,
                                int Pn, int C)
{
  int i = blockIdx.x * blockDim.x + threadIdx.x;
  if (i >= Bc * C) return;
  int b = i / C, c = i % C;
  float s = 0.f;
  for (int p = 0; p < Pn; ++p) s += x[((size_t)b * Pn + p) * C + c];
  feat[i] = s / (float)Pn;
}

__global__ void headcat_kernel(const float* __restrict__ feat, const float* __restrict__ cls,
                               const float* __restrict__ cv, float* __restrict__ h)
{
  int i = blockIdx.x * blockDim.x + threadIdx.x;
  if (i >= Bc * 400) return;
  int b = i / 400, j = i % 400;
  float v;
  if (j < 192)      v = feat[b * 192 + j];
  else if (j < 200) v = cls[b * 8 + (j - 192)];
  else              v = cv[b * 200 + (j - 200)];
  h[i] = v;
}

__global__ void outcat_kernel(const float* __restrict__ x1, const float* __restrict__ x2,
                              const float* __restrict__ x3, float* __restrict__ out)
{
  int i = blockIdx.x * blockDim.x + threadIdx.x;
  if (i >= Bc * 39) return;
  int b = i / 39, j = i % 39;
  float v;
  if (j < 12)       v = x1[b * 13 + j];
  else if (j < 24)  v = x2[b * 13 + (j - 12)];
  else if (j < 36)  v = x3[b * 13 + (j - 24)];
  else if (j == 36) v = x1[b * 13 + 12];
  else if (j == 37) v = x2[b * 13 + 12];
  else              v = x3[b * 13 + 12];
  out[i] = v;
}

// ============================================================================
// Host-side orchestration
// ============================================================================
struct Bump {
  char* base; size_t off;
  float* f(size_t n) { size_t o = (off + 255) & ~(size_t)255; off = o + n * 4; return (float*)(base + o); }
  int*   i(size_t n) { size_t o = (off + 255) & ~(size_t)255; off = o + n * 4; return (int*)(base + o); }
};

static void run_gemm(int act, const float* A, const float* W, const float* b,
                     float* C, long M, int N, int K, hipStream_t s)
{
  dim3 g((unsigned)CDIV(N, 32), (unsigned)CDIV(M, 32));
  dim3 blk(128);
  if (act == 0)      wmma_gemm_kernel<0><<<g, blk, 0, s>>>(A, W, b, C, (int)M, N, K);
  else if (act == 1) wmma_gemm_kernel<1><<<g, blk, 0, s>>>(A, W, b, C, (int)M, N, K);
  else               wmma_gemm_kernel<2><<<g, blk, 0, s>>>(A, W, b, C, (int)M, N, K);
}

static void run_bn(float* X, long Nrows, int C, const float* g, const float* be,
                   float* mbuf, float* vbuf, hipStream_t s)
{
  bn_stats_kernel<<<C, 256, 0, s>>>(X, mbuf, vbuf, Nrows, C);
  long tot = Nrows * C;
  bn_apply_kernel<<<(unsigned)CDIV(tot, 256), 256, 0, s>>>(X, mbuf, vbuf, g, be, Nrows, C);
}

static void run_knn(const float* pos, int* nbr, int Pn, int Kd, hipStream_t s)
{
  if (Kd == 8)       knn_kernel<8><<<Bc, 256, 0, s>>>(pos, nbr, Pn);
  else if (Kd == 24) knn_kernel<24><<<Bc, 256, 0, s>>>(pos, nbr, Pn);
  else               knn_kernel<32><<<Bc, 256, 0, s>>>(pos, nbr, Pn);
}

// Leaf order inside each pcnn param dict (JAX pytree: keys sorted):
// bn1(be,g) bn2(be,g) dw(b,w) fl(b,w) l1(b,w) l2(b,w)
// tbn1(be,g) tbn2(be,g) tbn3(be,g) tc1(b,w) tc2(b,w) tl(b,w)
struct XP {
  const float *bn1be,*bn1g,*bn2be,*bn2g,*dwb,*dww,*flb,*flw,*l1b,*l1w,*l2b,*l2w,
              *tbn1be,*tbn1g,*tbn2be,*tbn2g,*tbn3be,*tbn3g,*tc1b,*tc1w,*tc2b,*tc2w,*tlb,*tlw;
};
static XP xp_from(void* const* d_in, int base)
{
  auto f = [&](int i) { return (const float*)d_in[base + i]; };
  XP p;
  p.bn1be=f(0);  p.bn1g=f(1);  p.bn2be=f(2);  p.bn2g=f(3);
  p.dwb=f(4);    p.dww=f(5);   p.flb=f(6);    p.flw=f(7);
  p.l1b=f(8);    p.l1w=f(9);   p.l2b=f(10);   p.l2w=f(11);
  p.tbn1be=f(12);p.tbn1g=f(13);p.tbn2be=f(14);p.tbn2g=f(15);
  p.tbn3be=f(16);p.tbn3g=f(17);p.tc1b=f(18);  p.tc1w=f(19);
  p.tc2b=f(20);  p.tc2w=f(21); p.tlb=f(22);   p.tlw=f(23);
  return p;
}

static void run_xconv(const float* x, const float* pos, float* xout,
                      int Pn, int K, int dil, int Cd, int Cin, int dm, int Cout,
                      const XP& pp, Bump ws, unsigned seed, hipStream_t s)
{
  const long BP  = (long)Bc * Pn;
  const long BPK = BP * K;
  const int  Cm  = Cd + Cin;
  const int  KK  = K * K;
  const int  Kd  = K * dil;

  float* mbuf = ws.f(512);
  float* vbuf = ws.f(512);

  int* nbrF = ws.i(BP * Kd);
  run_knn(pos, nbrF, Pn, Kd, s);
  int* nbr = nbrF;
  if (dil > 1) {
    nbr = ws.i(BPK);
    dilate_kernel<<<(unsigned)CDIV(BPK, 256), 256, 0, s>>>(nbrF, nbr, BPK, K, Kd, seed);
  }
  float* rel = ws.f(BPK * 3);
  gather_rel_kernel<<<(unsigned)CDIV(BPK, 256), 256, 0, s>>>(pos, nbr, rel, Pn, K);

  // mlp1: h = BN(elu(rel @ l1)), x_star = BN(elu(h @ l2))
  float* h = ws.f(BPK * Cd);
  run_gemm(2, rel, pp.l1w, pp.l1b, h, BPK, Cd, 3, s);
  run_bn(h, BPK, Cd, pp.bn1g, pp.bn1be, mbuf, vbuf, s);
  float* xs = ws.f(BPK * Cd);
  run_gemm(2, h, pp.l2w, pp.l2b, xs, BPK, Cd, Cd, s);
  run_bn(xs, BPK, Cd, pp.bn2g, pp.bn2be, mbuf, vbuf, s);

  float* xcat = xs;
  if (Cin > 0) {
    xcat = ws.f(BPK * Cm);
    concat_kernel<<<(unsigned)CDIV(BPK * Cm, 256), 256, 0, s>>>(xs, x, nbr, xcat, Pn, K, Cd, Cin);
  }

  // mlp2: learned KxK X-transform
  float* t0 = ws.f(BP * KK);
  float* t1 = ws.f(BP * KK);
  run_gemm(2, rel, pp.tlw, pp.tlb, t0, BP, KK, 3 * K, s);      // elu fused
  run_bn(t0, BP, KK, pp.tbn1g, pp.tbn1be, mbuf, vbuf, s);
  tconv_kernel<<<(unsigned)CDIV(BP * KK, 256), 256, 0, s>>>(t0, pp.tc1w, pp.tc1b, t1, BP, K, 1);
  run_bn(t1, BP, KK, pp.tbn2g, pp.tbn2be, mbuf, vbuf, s);
  tconv_kernel<<<(unsigned)CDIV(BP * KK, 256), 256, 0, s>>>(t1, pp.tc2w, pp.tc2b, t0, BP, K, 0);
  run_bn(t0, BP, KK, pp.tbn3g, pp.tbn3be, mbuf, vbuf, s);      // T

  // x^T @ T, depthwise conv, final linear (+outer relu)
  float* y = ws.f(BP * (long)Cm * dm);
  xt_dw_kernel<<<(unsigned)CDIV(BP * Cm, 256), 256, 0, s>>>(xcat, t0, pp.dww, pp.dwb, y, BP, K, Cm, dm);
  run_gemm(1, y, pp.flw, pp.flb, xout, BP, Cout, Cm * dm, s);
}

extern "C" void kernel_launch(void* const* d_in, const int* in_sizes, int n_in,
                              void* d_out, int out_size, void* d_ws, size_t ws_size,
                              hipStream_t stream)
{
  (void)in_sizes; (void)n_in; (void)out_size; (void)ws_size;
  // Input order: pos, ctr, batch, then params flattened as a JAX pytree
  // (dict keys sorted): cls_w, lin1(b,w), lin2, lin3, lin_phi, lin_psi,
  // lin_theta, linctr, pcnn1[24 leaves], pcnn2[24], pcnn3[24].
  const float* pos0  = (const float*)d_in[0];   // [32,1024,3]
  const float* ctr   = (const float*)d_in[1];   // [32,3]
  const float* clsw  = (const float*)d_in[3];   // [3,8]
  const float* l1b   = (const float*)d_in[4];  const float* l1w  = (const float*)d_in[5];
  const float* l2b   = (const float*)d_in[6];  const float* l2w  = (const float*)d_in[7];
  const float* l3b   = (const float*)d_in[8];  const float* l3w  = (const float*)d_in[9];
  const float* lphib = (const float*)d_in[10]; const float* lphiw = (const float*)d_in[11];
  const float* lpsib = (const float*)d_in[12]; const float* lpsiw = (const float*)d_in[13];
  const float* ltheb = (const float*)d_in[14]; const float* lthew = (const float*)d_in[15];
  const float* lctrb = (const float*)d_in[16]; const float* lctrw = (const float*)d_in[17];
  XP p1 = xp_from(d_in, 18);
  XP p2 = xp_from(d_in, 42);
  XP p3 = xp_from(d_in, 66);
  float* out = (float*)d_out;

  Bump ws{(char*)d_ws, 0};
  // persistent buffers
  float* mp   = ws.f(Bc * 3);
  float* clsb = ws.f(Bc * 8);
  float* cvec = ws.f(Bc * 200);
  float* feat = ws.f(Bc * 192);
  float* hcat = ws.f(Bc * 400);
  float* hh1  = ws.f(Bc * 512);
  float* hh2  = ws.f(Bc * 256);
  float* hh3  = ws.f(Bc * 128);
  float* o1   = ws.f(Bc * 13);
  float* o2   = ws.f(Bc * 13);
  float* o3   = ws.f(Bc * 13);
  float* pos1 = ws.f((size_t)Bc * P1c * 3);
  float* pos2 = ws.f((size_t)Bc * P2c * 3);
  float* x0   = ws.f((size_t)Bc * P0 * 48);
  float* x1in = ws.f((size_t)Bc * P1c * 48);
  float* x1   = ws.f((size_t)Bc * P1c * 96);
  float* x2in = ws.f((size_t)Bc * P2c * 96);
  float* x2   = ws.f((size_t)Bc * P2c * 192);
  int*   idx1 = ws.i(Bc * P1c);
  int*   idx2 = ws.i(Bc * P2c);
  // everything after this point is per-layer scratch (Bump passed by value)

  // frozen classifier stub: one-hot(argmax(mean(pos) @ cls_w))
  meanpos_kernel<<<CDIV(Bc * 3, 64), 64, 0, stream>>>(pos0, mp);
  cls_kernel<<<1, Bc, 0, stream>>>(mp, clsw, clsb);

  // xconv 1: no input features, K=8, dil=1, Cd=32, dm=2, Cout=48
  run_xconv(nullptr, pos0, x0, P0, 8, 1, 32, 0, 2, 48, p1, ws, 0u, stream);
  fps_kernel<<<Bc, 256, 0, stream>>>(pos0, idx1, P0, P1c);
  take_kernel<<<(unsigned)CDIV((long)Bc * P1c * 48, 256), 256, 0, stream>>>(x0, idx1, x1in, P0, P1c, 48);
  take_kernel<<<(unsigned)CDIV((long)Bc * P1c * 3, 256), 256, 0, stream>>>(pos0, idx1, pos1, P0, P1c, 3);

  // xconv 2: K=12, dil=2, Cd=64, Cin=48, dm=1, Cout=96
  run_xconv(x1in, pos1, x1, P1c, 12, 2, 64, 48, 1, 96, p2, ws, 0x9E3779B9u, stream);
  fps_kernel<<<Bc, 256, 0, stream>>>(pos1, idx2, P1c, P2c);
  take_kernel<<<(unsigned)CDIV((long)Bc * P2c * 96, 256), 256, 0, stream>>>(x1, idx2, x2in, P1c, P2c, 96);
  take_kernel<<<(unsigned)CDIV((long)Bc * P2c * 3, 256), 256, 0, stream>>>(pos1, idx2, pos2, P1c, P2c, 3);

  // xconv 3: K=16, dil=2, Cd=128, Cin=96, dm=1, Cout=192
  run_xconv(x2in, pos2, x2, P2c, 16, 2, 128, 96, 1, 192, p3, ws, 0x85EBCA6Bu, stream);

  // head
  meanpool_kernel<<<CDIV(Bc * 192, 256), 256, 0, stream>>>(x2, feat, P2c, 192);
  run_gemm(1, ctr, lctrw, lctrb, cvec, Bc, 200, 3, stream);
  headcat_kernel<<<CDIV(Bc * 400, 256), 256, 0, stream>>>(feat, clsb, cvec, hcat);
  run_gemm(1, hcat, l1w, l1b, hh1, Bc, 512, 400, stream);
  run_gemm(1, hh1, l2w, l2b, hh2, Bc, 256, 512, stream);
  run_gemm(1, hh2, l3w, l3b, hh3, Bc, 128, 256, stream);
  run_gemm(0, hh3, lpsiw, lpsib, o1, Bc, 13, 128, stream);
  run_gemm(0, hh3, lthew, ltheb, o2, Bc, 13, 128, stream);
  run_gemm(0, hh3, lphiw, lphib, o3, Bc, 13, 128, stream);
  outcat_kernel<<<CDIV(Bc * 39, 64), 64, 0, stream>>>(o1, o2, o3, out);
}
